// SPVDownStage_86887188398531
// MI455X (gfx1250) — matrix operand order repaired and checked by reading, not compiled, not run on hardware
//
#include <hip/hip_runtime.h>
#include <hip/hip_bf16.h>

typedef __attribute__((ext_vector_type(16))) _Float16 v16h;
typedef __attribute__((ext_vector_type(8)))  float    v8f;

#define NVX  250000
#define NV1X 100000
#define NV2X 40000
#define NPX  500000

__device__ __forceinline__ float silu_f(float x) {
    return x * (1.0f / (1.0f + __expf(-x)));
}

// ---------------------------------------------------------------------------
// Column mean/var statistics: accumulate sum & sum-of-squares per column.
// ncols must be a power of two <= 64; blockDim.x = 256.
// ---------------------------------------------------------------------------
__global__ __launch_bounds__(256)
void k_colstats(const float* __restrict__ X, int nrows, int ncols,
                float* __restrict__ osum, float* __restrict__ osq)
{
    __shared__ float ssum[64], ssq[64];
    const int tid = threadIdx.x;
    if (tid < ncols) { ssum[tid] = 0.f; ssq[tid] = 0.f; }
    __syncthreads();

    const int col  = tid & (ncols - 1);
    const int rgrp = tid / ncols;
    const int rpb  = 256 / ncols;
    float s = 0.f, q = 0.f;
    for (long r = (long)blockIdx.x * rpb + rgrp; r < nrows;
         r += (long)gridDim.x * rpb) {
        float v = X[r * (long)ncols + col];
        s += v; q += v * v;
    }
    atomicAdd(&ssum[col], s);
    atomicAdd(&ssq[col], q);
    __syncthreads();
    if (tid < ncols) {
        atomicAdd(&osum[tid], ssum[tid]);
        atomicAdd(&osq[tid],  ssq[tid]);
    }
}

// ---------------------------------------------------------------------------
// eterm = silu(emb) @ we + be    (emb: 256, we: 256 x ni, ni <= 64)
// ---------------------------------------------------------------------------
__global__ __launch_bounds__(64)
void k_embterm(const float* __restrict__ emb, const float* __restrict__ we,
               const float* __restrict__ be, float* __restrict__ out, int ni)
{
    const int j = threadIdx.x;
    if (j >= ni) return;
    float acc = be[j];
    for (int i = 0; i < 256; ++i) {
        acc += silu_f(emb[i]) * we[i * ni + j];
    }
    out[j] = acc;
}

// ---------------------------------------------------------------------------
// Fused WMMA stage, fully specialized at compile time:
//   Out = [silu(BN(A)) | A/rowcnt | A] @ W + bias (+extra) (+resid)
//   optional scatter-add into segment sums; optional column stats of output.
// One wave = one 16-row tile; blockDim = 256 (8 waves -> 128 rows / block).
// Operands staged in LDS pre-swizzled into the CDNA5 WMMA fragment layouts:
//   A (16x32 f16): lane = m + 16*kh ; element e covers K = e+8*kh+(e>=8?8:0)
//   B (32x16 f16): lane = n + 16*kh ; element e covers K = e+8*kh+(e>=8?8:0)
//   C/D (16x16 f32): lane = n + 16*rh ; c[j] = D[j + 8*rh][n]
// so every fragment is one contiguous 32-byte LDS vector load.
// ---------------------------------------------------------------------------
template<int K, int N, bool USE_BN, bool USE_CNT, bool HAS_EXTRA,
         bool HAS_RESID, bool HAS_OUT, bool HAS_SEG, bool HAS_STAT>
__global__ __launch_bounds__(256)
void k_wmma_stage(const float* __restrict__ A, int nrows,
                  const float* __restrict__ bn_s, const float* __restrict__ bn_q,
                  float bn_inv_n,
                  const float* __restrict__ gamma, const float* __restrict__ beta,
                  const float* __restrict__ rowcnt,
                  const float* __restrict__ W, const float* __restrict__ bias,
                  const float* __restrict__ extra,
                  const float* __restrict__ resid,
                  float* __restrict__ Out,
                  const int* __restrict__ segidx,
                  float* __restrict__ segsum, float* __restrict__ segcnt,
                  float* __restrict__ colsum, float* __restrict__ colsq)
{
    constexpr int KSTEPS = K / 32;
    constexpr int NTILES = N / 16;

    __shared__ __align__(32) _Float16 sWf[K * N];
    __shared__ __align__(32) _Float16 sAf[8][KSTEPS * 512];
    __shared__ float sScale[K], sShift[K];
    __shared__ int   sSeg[8][16];

    const int tid  = threadIdx.x;
    const int wv   = tid >> 5;
    const int lane = tid & 31;

    if constexpr (USE_BN) {
        if (tid < K) {
            float m  = bn_s[tid] * bn_inv_n;
            float vv = bn_q[tid] * bn_inv_n - m * m;
            float rs = rsqrtf(vv + 1e-5f);
            float sc = gamma[tid] * rs;
            sScale[tid] = sc;
            sShift[tid] = beta[tid] - m * sc;
        }
    }

    // Stage weights into B-fragment layout.
    for (int i = tid; i < K * N; i += 256) {
        int e    = i & 15;
        int frag = i >> 4;
        int fl   = frag & 31;
        int tk   = frag >> 5;
        int ks   = tk & (KSTEPS - 1);
        int t    = tk / KSTEPS;
        int kk   = ks * 32 + (e & 7) + 8 * (fl >> 4) + ((e & 8) ? 16 : 0);
        int col  = t * 16 + (fl & 15);
        sWf[i] = (_Float16)W[kk * N + col];
    }
    __syncthreads();

    const long r0 = (long)blockIdx.x * 128 + (long)wv * 16;
    const bool fullTile = (r0 + 16 <= (long)nrows);

    if constexpr (HAS_SEG) {
        if (lane < 16) {
            long row = r0 + lane;
            int  si  = (row < nrows) ? segidx[row] : 0;
            sSeg[wv][lane] = si;
            if (row < nrows) atomicAdd(&segcnt[si], 1.0f);
        }
    }

    // Stage A tile (preprocessing applied) into the A-fragment layout.
    for (int i = lane; i < 16 * K; i += 32) {
        int  m   = i / K;
        int  k   = i - m * K;
        long row = r0 + m;
        float v  = 0.f;
        if (row < nrows) {
            v = A[row * (long)K + k];
            if constexpr (USE_CNT) v *= 1.0f / fmaxf(rowcnt[row], 1.0f);
            if constexpr (USE_BN)  v = silu_f(v * sScale[k] + sShift[k]);
        }
        int ks = k >> 5;
        int kl = k & 31;
        int kh = (kl >> 3) & 1;
        int e  = (kl & 7) + ((kl & 16) ? 8 : 0);
        int fl = m + 16 * kh;
        sAf[wv][((ks * 32 + fl) << 4) + e] = (_Float16)v;
    }
    __syncthreads();

    v16h afrag[KSTEPS];
#pragma unroll
    for (int ks = 0; ks < KSTEPS; ++ks)
        afrag[ks] = *reinterpret_cast<const v16h*>(
            &sAf[wv][(ks * 32 + lane) << 4]);

    const int bn_ = lane & 15;
    const int rh  = lane >> 4;

#pragma unroll
    for (int t = 0; t < NTILES; ++t) {
        v8f c = {0.f, 0.f, 0.f, 0.f, 0.f, 0.f, 0.f, 0.f};
#pragma unroll
        for (int ks = 0; ks < KSTEPS; ++ks) {
            v16h bfrag = *reinterpret_cast<const v16h*>(
                &sWf[((t * KSTEPS + ks) * 32 + lane) << 4]);
            c = __builtin_amdgcn_wmma_f32_16x16x32_f16(
                    false, afrag[ks], false, bfrag, (short)0, c, false, false);
        }
        const int col = t * 16 + bn_;
        float badd = bias[col];
        if constexpr (HAS_EXTRA) badd += extra[col];
        float csum = 0.f, csq = 0.f;
        if (fullTile) {
#pragma unroll
            for (int j = 0; j < 8; ++j) {
                long row = r0 + j + 8 * rh;
                float v = c[j] + badd;
                if constexpr (HAS_RESID) v += resid[row * (long)N + col];
                if constexpr (HAS_OUT)   Out[row * (long)N + col] = v;
                if constexpr (HAS_SEG)
                    atomicAdd(&segsum[(long)sSeg[wv][j + 8 * rh] * N + col], v);
                if constexpr (HAS_STAT) { csum += v; csq += v * v; }
            }
        } else {
#pragma unroll
            for (int j = 0; j < 8; ++j) {
                long row = r0 + j + 8 * rh;
                if (row < nrows) {
                    float v = c[j] + badd;
                    if constexpr (HAS_RESID) v += resid[row * (long)N + col];
                    if constexpr (HAS_OUT)   Out[row * (long)N + col] = v;
                    if constexpr (HAS_SEG)
                        atomicAdd(&segsum[(long)sSeg[wv][j + 8 * rh] * N + col], v);
                    if constexpr (HAS_STAT) { csum += v; csq += v * v; }
                }
            }
        }
        if constexpr (HAS_STAT) {
            atomicAdd(&colsum[col], csum);
            atomicAdd(&colsq[col],  csq);
        }
    }
}

// ---------------------------------------------------------------------------
// voxel_to_point: z1[p,:] = sum_k nbr_w[p,k] * f[nbr_idx[p,k], :]   (C = 128)
// One block per point; blockDim = 128.  f (20 MB) is L2-resident.
// ---------------------------------------------------------------------------
__global__ __launch_bounds__(128)
void k_gather(const float* __restrict__ f, const int* __restrict__ nidx,
              const float* __restrict__ nw, float* __restrict__ z1, int np)
{
    const int p = blockIdx.x;
    if (p >= np) return;
    __shared__ int   si[8];
    __shared__ float sw[8];
    if (threadIdx.x < 8) {
        si[threadIdx.x] = nidx[(long)p * 8 + threadIdx.x];
        sw[threadIdx.x] = nw[(long)p * 8 + threadIdx.x];
    }
    __syncthreads();
    const int c = threadIdx.x;
    float acc = 0.f;
#pragma unroll
    for (int k = 0; k < 8; ++k)
        acc += sw[k] * f[(long)si[k] * 128 + c];
    z1[(long)p * 128 + c] = acc;
}

// ---------------------------------------------------------------------------
// x_out[v,c] = xsum[v,c] / max(xcnt[v], 1)
// ---------------------------------------------------------------------------
__global__ __launch_bounds__(256)
void k_finalize(const float* __restrict__ s, const float* __restrict__ cnt,
                float* __restrict__ out, int nv, int nc)
{
    long i = (long)blockIdx.x * 256 + threadIdx.x;
    long tot = (long)nv * nc;
    if (i >= tot) return;
    long v = i / nc;
    out[i] = s[i] / fmaxf(cnt[v], 1.0f);
}

// ---------------------------------------------------------------------------
extern "C" void kernel_launch(void* const* d_in, const int* in_sizes, int n_in,
                              void* d_out, int out_size, void* d_ws, size_t ws_size,
                              hipStream_t stream)
{
    const float* x     = (const float*)d_in[0];
    const float* z     = (const float*)d_in[1];
    const float* emb   = (const float*)d_in[2];
    const int*   pidx1 = (const int*)d_in[3];
    const int*   pidx2 = (const int*)d_in[4];
    const int*   nidx  = (const int*)d_in[5];
    const float* nw    = (const float*)d_in[6];
    const int*   pvidx = (const int*)d_in[7];
    // block 0 params (insertion order of the reference dict)
    const float* b0_bn1g = (const float*)d_in[8];
    const float* b0_bn1b = (const float*)d_in[9];
    const float* b0_w1   = (const float*)d_in[10];
    const float* b0_b1   = (const float*)d_in[11];
    const float* b0_we   = (const float*)d_in[12];
    const float* b0_be   = (const float*)d_in[13];
    const float* b0_bn2g = (const float*)d_in[14];
    const float* b0_bn2b = (const float*)d_in[15];
    const float* b0_w2   = (const float*)d_in[16];
    const float* b0_b2   = (const float*)d_in[17];
    const float* b0_wd   = (const float*)d_in[18];
    const float* b0_bd   = (const float*)d_in[19];
    // block 1 params
    const float* b1_bn1g = (const float*)d_in[20];
    const float* b1_bn1b = (const float*)d_in[21];
    const float* b1_w1   = (const float*)d_in[22];
    const float* b1_b1   = (const float*)d_in[23];
    const float* b1_we   = (const float*)d_in[24];
    const float* b1_be   = (const float*)d_in[25];
    const float* b1_bn2g = (const float*)d_in[26];
    const float* b1_bn2b = (const float*)d_in[27];
    const float* b1_w2   = (const float*)d_in[28];
    const float* b1_b2   = (const float*)d_in[29];
    const float* b1_wd   = (const float*)d_in[30];
    const float* b1_bd   = (const float*)d_in[31];
    // point block
    const float* pb_g    = (const float*)d_in[32];
    const float* pb_b    = (const float*)d_in[33];
    const float* pb_w    = (const float*)d_in[34];
    const float* pb_bias = (const float*)d_in[35];

    float* w = (float*)d_ws;
    size_t o = 0;
    float* sx_s = w + o; o += 32;  float* sx_q = w + o; o += 32;
    float* s2_s = w + o; o += 32;  float* s2_q = w + o; o += 32;
    float* s3_s = w + o; o += 64;  float* s3_q = w + o; o += 64;
    float* s4_s = w + o; o += 64;  float* s4_q = w + o; o += 64;
    float* sz_s = w + o; o += 32;  float* sz_q = w + o; o += 32;
    float* et1  = w + o; o += 32;  float* et2  = w + o; o += 64;
    const size_t statFloats = o;                    // 544
    float* h1   = w + o; o += (size_t)NVX * 32;
    float* seg1 = w + o; o += (size_t)NV1X * 32;
    float* cnt1 = w + o; o += (size_t)NV1X;
    float* f1d  = w + o; o += (size_t)NV1X * 64;
    float* h2   = w + o; o += (size_t)NV1X * 64;
    float* seg2 = w + o; o += (size_t)NV2X * 64;
    float* cnt2 = w + o; o += (size_t)NV2X;
    float* f2d  = w + o; o += (size_t)NV2X * 128;
    float* xsum = w + o; o += (size_t)NV2X * 128;
    float* xcnt = w + o; o += (size_t)NV2X;

    float* xout = (float*)d_out;                      // NV2 x 128
    float* z1   = xout + (size_t)NV2X * 128;          // NP  x 128

    // Zero accumulators (capturable async memsets, stream-ordered).
    hipMemsetAsync(w, 0, statFloats * sizeof(float), stream);
    hipMemsetAsync(seg1, 0, (size_t)NV1X * 33 * sizeof(float), stream);
    hipMemsetAsync(seg2, 0, (size_t)NV2X * 65 * sizeof(float), stream);
    hipMemsetAsync(xsum, 0, (size_t)NV2X * 129 * sizeof(float), stream);

    // Input statistics for BN1(x) and BN(z).
    k_colstats<<<dim3(512), dim3(256), 0, stream>>>(x, NVX, 32, sx_s, sx_q);
    k_colstats<<<dim3(512), dim3(256), 0, stream>>>(z, NPX, 32, sz_s, sz_q);

    // Time-embedding terms.
    k_embterm<<<dim3(1), dim3(64), 0, stream>>>(emb, b0_we, b0_be, et1, 32);
    k_embterm<<<dim3(1), dim3(64), 0, stream>>>(emb, b1_we, b1_be, et2, 64);

    const float invNV  = 1.0f / (float)NVX;
    const float invNV1 = 1.0f / (float)NV1X;
    const float invNP  = 1.0f / (float)NPX;

    // ---- Block 1 (ni=32 -> nf=64, pool NV -> NV1) ----
    // C: h1 = silu(BN1(x)) @ w1 + b1 + et1 ; stats of h1.
    k_wmma_stage<32, 32, true, false, true, false, true, false, true>
        <<<dim3((NVX + 127) / 128), dim3(256), 0, stream>>>(
        x, NVX, sx_s, sx_q, invNV, b0_bn1g, b0_bn1b, nullptr,
        b0_w1, b0_b1, et1, nullptr, h1,
        nullptr, nullptr, nullptr, s2_s, s2_q);
    // D: f = x + silu(BN2(h1)) @ w2 + b2 ; scatter into seg1 via pool_idx1.
    k_wmma_stage<32, 32, true, false, false, true, false, true, false>
        <<<dim3((NVX + 127) / 128), dim3(256), 0, stream>>>(
        h1, NVX, s2_s, s2_q, invNV, b0_bn2g, b0_bn2b, nullptr,
        b0_w2, b0_b2, nullptr, x, nullptr,
        pidx1, seg1, cnt1, nullptr, nullptr);
    // E: f1d = segmean(seg1) @ wd + bd ; stats for block 2's BN1.
    k_wmma_stage<32, 64, false, true, false, false, true, false, true>
        <<<dim3((NV1X + 127) / 128), dim3(256), 0, stream>>>(
        seg1, NV1X, nullptr, nullptr, 0.f, nullptr, nullptr, cnt1,
        b0_wd, b0_bd, nullptr, nullptr, f1d,
        nullptr, nullptr, nullptr, s3_s, s3_q);

    // ---- Block 2 (ni=64 -> nf=128, pool NV1 -> NV2) ----
    k_wmma_stage<64, 64, true, false, true, false, true, false, true>
        <<<dim3((NV1X + 127) / 128), dim3(256), 0, stream>>>(
        f1d, NV1X, s3_s, s3_q, invNV1, b1_bn1g, b1_bn1b, nullptr,
        b1_w1, b1_b1, et2, nullptr, h2,
        nullptr, nullptr, nullptr, s4_s, s4_q);
    k_wmma_stage<64, 64, true, false, false, true, false, true, false>
        <<<dim3((NV1X + 127) / 128), dim3(256), 0, stream>>>(
        h2, NV1X, s4_s, s4_q, invNV1, b1_bn2g, b1_bn2b, nullptr,
        b1_w2, b1_b2, nullptr, f1d, nullptr,
        pidx2, seg2, cnt2, nullptr, nullptr);
    k_wmma_stage<64, 128, false, true, false, false, true, false, false>
        <<<dim3((NV2X + 127) / 128), dim3(256), 0, stream>>>(
        seg2, NV2X, nullptr, nullptr, 0.f, nullptr, nullptr, cnt2,
        b1_wd, b1_bd, nullptr, nullptr, f2d,
        nullptr, nullptr, nullptr, nullptr, nullptr);

    // ---- voxel_to_point trilinear gather ----
    k_gather<<<dim3(NPX), dim3(128), 0, stream>>>(f2d, nidx, nw, z1, NPX);

    // ---- PointBlock + point_to_voxel scatter sums ----
    k_wmma_stage<32, 128, true, false, false, true, true, true, false>
        <<<dim3((NPX + 127) / 128), dim3(256), 0, stream>>>(
        z, NPX, sz_s, sz_q, invNP, pb_g, pb_b, nullptr,
        pb_w, pb_bias, nullptr, z1, z1,
        pvidx, xsum, xcnt, nullptr, nullptr);

    // ---- x_out = xsum / max(xcnt, 1) ----
    long tot = (long)NV2X * 128;
    k_finalize<<<dim3((unsigned)((tot + 255) / 256)), dim3(256), 0, stream>>>(
        xsum, xcnt, xout, NV2X, 128);

    (void)in_sizes; (void)n_in; (void)out_size; (void)ws_size;
}
